// Head_3040836846186
// MI455X (gfx1250) — matrix operand order repaired and checked by reading, not compiled, not run on hardware
//
#include <hip/hip_runtime.h>
#include <hip/hip_bf16.h>

typedef __attribute__((ext_vector_type(16))) _Float16 v16h;
typedef __attribute__((ext_vector_type(8)))  float    v8f;

#define B_SZ 512
#define T_SZ 256
#define C_SZ 256
#define H_SZ 64

#define WT_PITCH 264   // fp16 elems per W-transposed row (256 + 8 pad); 528B = 33*16
#define KS_PITCH 72    // fp16 elems per K row (64 + 8 pad); 144B = 9*16
#define VT_PITCH 264   // fp16 elems per V-transposed row
#define P_PITCH  264   // fp16 elems per P row

static __device__ __forceinline__ v8f wmma_f16(v16h a, v16h b, v8f c) {
    // D = A(16x32 f16) * B(32x16 f16) + C(16x16 f32)
    return __builtin_amdgcn_wmma_f32_16x16x32_f16(false, a, false, b, (short)0, c,
                                                  false, false);
}

// build a v16h fragment from two 16-byte chunks (generic/LDS/global pointers)
static __device__ __forceinline__ v16h ld_frag2(const _Float16* p0, const _Float16* p1) {
    v16h r;
    ((uint4*)&r)[0] = *(const uint4*)p0;
    ((uint4*)&r)[1] = *(const uint4*)p1;
    return r;
}

static __device__ __forceinline__ v16h cvt16(float4 f0, float4 f1, float4 f2, float4 f3) {
    v16h a;
    a[0]=(_Float16)f0.x;  a[1]=(_Float16)f0.y;  a[2]=(_Float16)f0.z;  a[3]=(_Float16)f0.w;
    a[4]=(_Float16)f1.x;  a[5]=(_Float16)f1.y;  a[6]=(_Float16)f1.z;  a[7]=(_Float16)f1.w;
    a[8]=(_Float16)f2.x;  a[9]=(_Float16)f2.y;  a[10]=(_Float16)f2.z; a[11]=(_Float16)f2.w;
    a[12]=(_Float16)f3.x; a[13]=(_Float16)f3.y; a[14]=(_Float16)f3.z; a[15]=(_Float16)f3.w;
    return a;
}

// CDNA5 async copy: global -> LDS without VGPR staging, tracked by ASYNCcnt.
// lds_byte_off = low 32 bits of a generic LDS pointer (aperture puts the LDS
// byte offset in addr[31:0]).
static __device__ __forceinline__ void async_b128(unsigned int lds_byte_off,
                                                  const void* gaddr) {
    asm volatile("global_load_async_to_lds_b128 %0, %1, off"
                 :: "v"(lds_byte_off), "v"(gaddr) : "memory");
}
static __device__ __forceinline__ void wait_async0() {
    asm volatile("s_wait_asynccnt 0" ::: "memory");
}
static __device__ __forceinline__ unsigned int lds_off(const void* p) {
    return (unsigned int)(unsigned long long)p;
}

// ---------------------------------------------------------------------------
// Kernel 1: Q = xWq, K = xWk, Vt = (xWv)^T   (fp32 in, fp16 out to scratch)
// one block per batch, 8 waves, each wave owns 2 row-tiles of 16 rows
// ---------------------------------------------------------------------------
__global__ __launch_bounds__(256) void qkv_proj(
    const float* __restrict__ x,  const float* __restrict__ Wq,
    const float* __restrict__ Wk, const float* __restrict__ Wv,
    _Float16* __restrict__ Qg, _Float16* __restrict__ Kg, _Float16* __restrict__ Vtg)
{
    __shared__ _Float16 Wt[3][H_SZ * WT_PITCH];   // W transposed [h][c], fp16

    const int tid = threadIdx.x;
    const int b   = blockIdx.x;

    // stage weights: Wt[p][h][c] = W_p[c][h], fp32 -> fp16 (needs conversion,
    // so this one stays on the VALU path)
    const float* Ws[3] = {Wq, Wk, Wv};
    for (int p = 0; p < 3; ++p) {
        const float* W = Ws[p];
        #pragma unroll
        for (int i = 0; i < (C_SZ * H_SZ) / 256; ++i) {
            int idx = i * 256 + tid;           // over C*H
            int c = idx >> 6, h = idx & 63;
            Wt[p][h * WT_PITCH + c] = (_Float16)W[idx];
        }
    }
    __syncthreads();

    const int wave = tid >> 5, lane = tid & 31;
    const int half = lane >> 4, ln = lane & 15;

    for (int rt = 0; rt < 2; ++rt) {
        const int trow = (wave * 2 + rt) * 16;
        const float* xr = x + ((size_t)b * T_SZ + trow + ln) * C_SZ;

        // A fragments of x: 16x32 f16, K = c.  Lane layout:
        //   lanes 0-15: K = k0..k0+7 (elems 0-7), k0+16..k0+23 (elems 8-15)
        //   lanes 16-31: shifted by +8
        v16h af[8];
        #pragma unroll
        for (int kk = 0; kk < 8; ++kk) {
            const int c0 = kk * 32 + half * 8;
            float4 f0 = *(const float4*)(xr + c0);
            float4 f1 = *(const float4*)(xr + c0 + 4);
            float4 f2 = *(const float4*)(xr + c0 + 16);
            float4 f3 = *(const float4*)(xr + c0 + 20);
            af[kk] = cvt16(f0, f1, f2, f3);
        }

        #pragma unroll
        for (int p = 0; p < 3; ++p) {
            #pragma unroll
            for (int nt = 0; nt < 4; ++nt) {     // H tile: hbase = nt*16
                v8f acc = {};
                #pragma unroll
                for (int kk = 0; kk < 8; ++kk) {
                    // B fragment 32x16: lane = column (h), halves split K
                    const _Float16* bp = &Wt[p][(nt * 16 + ln) * WT_PITCH
                                                + kk * 32 + half * 8];
                    acc = wmma_f16(af[kk], ld_frag2(bp, bp + 16), acc);
                }
                // C layout: VGPR r -> row trow + r + half*8, col nt*16 + ln
                #pragma unroll
                for (int r = 0; r < 8; ++r) {
                    const int row = trow + r + half * 8;
                    const int col = nt * 16 + ln;
                    _Float16 v = (_Float16)acc[r];
                    if (p == 0)
                        Qg[((size_t)b * T_SZ + row) * H_SZ + col] = v;
                    else if (p == 1)
                        Kg[((size_t)b * T_SZ + row) * H_SZ + col] = v;
                    else
                        Vtg[((size_t)b * H_SZ + col) * T_SZ + row] = v;
                }
            }
        }
    }
}

// ---------------------------------------------------------------------------
// Kernel 2: causal softmax(Q K^T * C^-0.5) @ V, one block per batch
// ---------------------------------------------------------------------------
__global__ __launch_bounds__(256) void attention(
    const _Float16* __restrict__ Qg, const _Float16* __restrict__ Kg,
    const _Float16* __restrict__ Vtg, float* __restrict__ out)
{
    __shared__ _Float16 Ks [T_SZ * KS_PITCH];      // K[b]  row-major [s][h]
    __shared__ _Float16 Vts[H_SZ * VT_PITCH];      // V[b]^T          [h][s]
    __shared__ _Float16 Ps [8][16 * P_PITCH];      // per-wave P pane [t][s]

    const int tid = threadIdx.x;
    const int b   = blockIdx.x;

    // stage K[b] and Vt[b] into LDS via CDNA5 async copies (no VGPR staging)
    {
        // K: one 64-fp16 row per thread = 8x b128 (row pitch padded in LDS)
        const _Float16* ksrc = Kg + (size_t)b * T_SZ * H_SZ + (size_t)tid * H_SZ;
        const unsigned int kdst = lds_off(Ks + tid * KS_PITCH);
        #pragma unroll
        for (int i = 0; i < 8; ++i)
            async_b128(kdst + i * 16, ksrc + i * 8);

        // Vt: 64 rows x 256 fp16; 4 threads per row, 8x b128 each
        const int vrow = tid >> 2, q = tid & 3;
        const _Float16* vsrc = Vtg + (size_t)b * H_SZ * T_SZ
                                   + (size_t)vrow * T_SZ + q * 64;
        const unsigned int vdst = lds_off(Vts + vrow * VT_PITCH + q * 64);
        #pragma unroll
        for (int i = 0; i < 8; ++i)
            async_b128(vdst + i * 16, vsrc + i * 8);

        wait_async0();
    }
    __syncthreads();

    const int wave = tid >> 5, lane = tid & 31;
    const int half = lane >> 4, ln = lane & 15;
    const float scale = 0.0625f;                  // C^-0.5 = 256^-0.5
    _Float16* pw = Ps[wave];

    for (int rt = 0; rt < 2; ++rt) {
        const int trow = (wave * 2 + rt) * 16;

        // Q A-fragments (16x32 over h), straight from fp16 scratch
        const _Float16* qr = Qg + ((size_t)b * T_SZ + trow + ln) * H_SZ;
        v16h qf[2];
        #pragma unroll
        for (int kk = 0; kk < 2; ++kk) {
            const int c0 = kk * 32 + half * 8;
            qf[kk] = ld_frag2(qr + c0, qr + c0 + 16);
        }

        // S = Q K^T : 16 tiles of 16x16, fp32 accumulators
        v8f s[16];
        #pragma unroll
        for (int st = 0; st < 16; ++st) {
            v8f acc = {};
            #pragma unroll
            for (int kk = 0; kk < 2; ++kk) {
                const _Float16* bp = Ks + (st * 16 + ln) * KS_PITCH
                                        + kk * 32 + half * 8;
                acc = wmma_f16(qf[kk], ld_frag2(bp, bp + 16), acc);
            }
            s[st] = acc;
        }

        // causal softmax per row; columns for this lane: st*16 + ln
        #pragma unroll
        for (int r = 0; r < 8; ++r) {
            const int row_t = trow + r + half * 8;
            float vals[16];
            float m = -3.0e38f;
            #pragma unroll
            for (int st = 0; st < 16; ++st) {
                const int col = st * 16 + ln;
                float v = s[st][r] * scale;
                v = (col <= row_t) ? v : -3.0e38f;
                vals[st] = v;
                m = fmaxf(m, v);
            }
            #pragma unroll
            for (int off = 1; off < 16; off <<= 1)
                m = fmaxf(m, __shfl_xor(m, off, 32));   // 16-lane halves hold disjoint cols
            float sum = 0.f;
            #pragma unroll
            for (int st = 0; st < 16; ++st) {
                float e = __expf(vals[st] - m);
                e = (vals[st] > -1.0e38f) ? e : 0.f;
                vals[st] = e;
                sum += e;
            }
            #pragma unroll
            for (int off = 1; off < 16; off <<= 1)
                sum += __shfl_xor(sum, off, 32);
            const float inv = 1.0f / sum;
            #pragma unroll
            for (int st = 0; st < 16; ++st)
                pw[(r + half * 8) * P_PITCH + st * 16 + ln] =
                    (_Float16)(vals[st] * inv);
        }
        // P pane is private to this wave; LDS ops are in-order within a wave.

        // out = P @ V : 4 column tiles, K = s (8 steps of 32)
        #pragma unroll
        for (int nt = 0; nt < 4; ++nt) {
            v8f acc = {};
            #pragma unroll
            for (int kk = 0; kk < 8; ++kk) {
                const int c0 = kk * 32 + half * 8;
                const _Float16* pr = pw + ln * P_PITCH;
                v16h afr = ld_frag2(pr + c0, pr + c0 + 16);
                const _Float16* bp = Vts + (nt * 16 + ln) * VT_PITCH
                                         + kk * 32 + half * 8;
                acc = wmma_f16(afr, ld_frag2(bp, bp + 16), acc);
            }
            #pragma unroll
            for (int r = 0; r < 8; ++r) {
                const int row = trow + r + half * 8;
                out[((size_t)b * T_SZ + row) * H_SZ + nt * 16 + ln] = acc[r];
            }
        }
    }
}

extern "C" void kernel_launch(void* const* d_in, const int* in_sizes, int n_in,
                              void* d_out, int out_size, void* d_ws, size_t ws_size,
                              hipStream_t stream) {
    const float* x  = (const float*)d_in[0];
    const float* Wq = (const float*)d_in[1];
    const float* Wk = (const float*)d_in[2];
    const float* Wv = (const float*)d_in[3];
    float* out = (float*)d_out;

    // fp16 scratch: Q[B,T,H], K[B,T,H], Vt[B,H,T] = 3 * 16 MB
    const size_t per = (size_t)B_SZ * T_SZ * H_SZ;
    _Float16* Qg  = (_Float16*)d_ws;
    _Float16* Kg  = Qg + per;
    _Float16* Vtg = Kg + per;

    qkv_proj<<<dim3(B_SZ), dim3(256), 0, stream>>>(x, Wq, Wk, Wv, Qg, Kg, Vtg);
    attention<<<dim3(B_SZ), dim3(256), 0, stream>>>(Qg, Kg, Vtg, out);
}
